// GINEncoder_12412455485739
// MI455X (gfx1250) — matrix operand order repaired and checked by reading, not compiled, not run on hardware
//
#include <hip/hip_runtime.h>
#include <hip/hip_bf16.h>

#define D_  64
#define HID 32

typedef __attribute__((ext_vector_type(16))) _Float16 v16h;
typedef __attribute__((ext_vector_type(8)))  float    v8f;
typedef _Float16 f16;

// ---------------- WMMA fragment loaders (LDS -> VGPR) ----------------
// A: 16x32 f16 tile of a row-major 16x64 LDS tile.
// Lanes 0-15: row = lane, elems 0..7 = K kb+0..7,  elems 8..15 = K kb+16..23
// Lanes 16-31: row = lane-16, elems 0..7 = K kb+8..15, elems 8..15 = K kb+24..31
__device__ __forceinline__ v16h ldsA(const f16* tile, int kb, int lane) {
    int row = lane & 15, hi = (lane >> 4) & 1;
    const f16* p = tile + row * D_ + kb + hi * 8;
    v16h a;
#pragma unroll
    for (int i = 0; i < 8; ++i) a[i] = p[i];
    p += 16;
#pragma unroll
    for (int i = 0; i < 8; ++i) a[8 + i] = p[i];
    return a;
}

// B: 32x16 f16 (KxN). Weights stored TRANSPOSED in LDS: wt[out][in], 64x64.
// Lanes 0-15: col = nt*16+lane, elems 0..7 = K kb+0..7, elems 8..15 = K kb+16..23 ; lanes 16-31: col = nt*16+lane-16, elems 0..7 = K kb+8..15, elems 8..15 = K kb+24..31
__device__ __forceinline__ v16h ldsB(const f16* wt, int nt, int kb, int lane) {
    int col = (nt << 4) + (lane & 15), hi = (lane >> 4) & 1;
    const f16* p = wt + col * D_ + kb + hi * 8;
    v16h b;
#pragma unroll
    for (int i = 0; i < 8; ++i) { b[i] = p[i]; b[8 + i] = p[16 + i]; }
    return b;
}

// ---------------- order-preserving float<->uint encode for atomic max ----------------
__device__ __forceinline__ unsigned encf(float x) {
    unsigned b = __float_as_uint(x);
    return (b & 0x80000000u) ? ~b : (b | 0x80000000u);
}
__device__ __forceinline__ float decf(unsigned u) {
    return __uint_as_float((u & 0x80000000u) ? (u & 0x7FFFFFFFu) : ~u);
}

// ---------------- weight convert: f32 [l][in][out] -> f16 transposed [l][out][in] ----------------
__global__ __launch_bounds__(256) void k_convert(const float* __restrict__ W1,
                                                 const float* __restrict__ W2,
                                                 unsigned short* __restrict__ w1t,
                                                 unsigned short* __restrict__ w2t,
                                                 int total) {
    int idx = blockIdx.x * 256 + threadIdx.x;
    if (idx >= total) return;
    int l = idx >> 12;          // / (64*64)
    int rem = idx & 4095;
    int o = rem >> 6, i = rem & 63;
    ((f16*)w1t)[idx] = (f16)W1[(l << 12) + i * D_ + o];
    ((f16*)w2t)[idx] = (f16)W2[(l << 12) + i * D_ + o];
}

// ---------------- edge scatter: agg[dst] += x[src], float atomics ----------------
__global__ __launch_bounds__(256) void k_scatter(const float* __restrict__ x,
                                                 const int* __restrict__ src,
                                                 const int* __restrict__ dst,
                                                 float* __restrict__ agg, int nE) {
    int idx = blockIdx.x * 256 + threadIdx.x;
    int e = idx >> 4;
    if (e >= nE) return;
    int q = idx & 15;
    int s = src[e], d = dst[e];
    float4 v = ((const float4*)(x + (size_t)s * D_))[q];
    float* o = agg + (size_t)d * D_ + q * 4;
    unsafeAtomicAdd(o + 0, v.x);
    unsafeAtomicAdd(o + 1, v.y);
    unsafeAtomicAdd(o + 2, v.z);
    unsafeAtomicAdd(o + 3, v.w);
}

// ---------------- fused GIN update: h=(1+eps)x+agg -> MLP via WMMA -> LayerNorm ----------------
// 128 threads = 4 waves; each wave owns a 16-node tile. 16 v_wmma per wave.
__global__ __launch_bounds__(128) void k_gin_mlp(const float* __restrict__ xin,
                                                 const float* __restrict__ agg,
                                                 const unsigned short* __restrict__ w1t_u,
                                                 const unsigned short* __restrict__ w2t_u,
                                                 const float* __restrict__ b1,
                                                 const float* __restrict__ b2,
                                                 const float* __restrict__ eps,
                                                 const float* __restrict__ lng,
                                                 const float* __restrict__ lnb,
                                                 float* __restrict__ xout,
                                                 int layer, int nN) {
    __shared__ f16 sW1[D_ * D_];
    __shared__ f16 sW2[D_ * D_];
    __shared__ float sB1[D_], sB2[D_], sG[D_], sBt[D_];
    __shared__ f16 sH [4][16 * D_];
    __shared__ f16 sH2[4][16 * D_];
    __shared__ float sO[4][16 * D_];

    int tid = threadIdx.x;
    // stage weights (uint4 = 8 halves per load)
    {
        const uint4* g1 = (const uint4*)(w1t_u + (size_t)layer * D_ * D_);
        const uint4* g2 = (const uint4*)(w2t_u + (size_t)layer * D_ * D_);
        uint4* s1 = (uint4*)sW1;
        uint4* s2 = (uint4*)sW2;
        for (int i = tid; i < D_ * D_ / 8; i += 128) { s1[i] = g1[i]; s2[i] = g2[i]; }
        if (tid < D_) {
            sB1[tid] = b1[layer * D_ + tid];
            sB2[tid] = b2[layer * D_ + tid];
            sG [tid] = lng[layer * D_ + tid];
            sBt[tid] = lnb[layer * D_ + tid];
        }
    }
    float epv = 1.0f + eps[layer];

    int wave = tid >> 5, lane = tid & 31;
    int node0 = blockIdx.x * 64 + wave * 16;

    // build h tile in f16: lane covers row = lane>>1, cols (lane&1)*32..+31
    {
        int row = lane >> 1, cb = (lane & 1) * 32;
        int n = node0 + row;
        if (n >= nN) n = nN - 1;  // clamp; OOB rows never stored
        const float4* px = (const float4*)(xin + (size_t)n * D_ + cb);
        const float4* pa = (const float4*)(agg + (size_t)n * D_ + cb);
        f16* hr = &sH[wave][row * D_ + cb];
#pragma unroll
        for (int q = 0; q < 8; ++q) {
            float4 xv = px[q], av = pa[q];
            hr[q * 4 + 0] = (f16)(epv * xv.x + av.x);
            hr[q * 4 + 1] = (f16)(epv * xv.y + av.y);
            hr[q * 4 + 2] = (f16)(epv * xv.z + av.z);
            hr[q * 4 + 3] = (f16)(epv * xv.w + av.w);
        }
    }
    __syncthreads();

    int colc = lane & 15, hi = (lane >> 4) & 1;

    // GEMM1: sH @ W1 + b1, SiLU -> sH2
    {
        v16h A0 = ldsA(sH[wave], 0, lane);
        v16h A1 = ldsA(sH[wave], 32, lane);
#pragma unroll
        for (int nt = 0; nt < 4; ++nt) {
            v8f c;
            float bv = sB1[nt * 16 + colc];
#pragma unroll
            for (int r = 0; r < 8; ++r) c[r] = bv;
            c = __builtin_amdgcn_wmma_f32_16x16x32_f16(false, A0, false, ldsB(sW1, nt, 0, lane),
                                                       (short)0, c, false, false);
            c = __builtin_amdgcn_wmma_f32_16x16x32_f16(false, A1, false, ldsB(sW1, nt, 32, lane),
                                                       (short)0, c, false, false);
#pragma unroll
            for (int r = 0; r < 8; ++r) {
                float v = c[r];
                v = v * (1.0f / (1.0f + __expf(-v)));  // SiLU
                sH2[wave][(r + hi * 8) * D_ + nt * 16 + colc] = (f16)v;
            }
        }
    }
    __syncthreads();

    // GEMM2: sH2 @ W2 + b2 -> sO (f32, pre-layernorm)
    {
        v16h A0 = ldsA(sH2[wave], 0, lane);
        v16h A1 = ldsA(sH2[wave], 32, lane);
#pragma unroll
        for (int nt = 0; nt < 4; ++nt) {
            v8f c;
            float bv = sB2[nt * 16 + colc];
#pragma unroll
            for (int r = 0; r < 8; ++r) c[r] = bv;
            c = __builtin_amdgcn_wmma_f32_16x16x32_f16(false, A0, false, ldsB(sW2, nt, 0, lane),
                                                       (short)0, c, false, false);
            c = __builtin_amdgcn_wmma_f32_16x16x32_f16(false, A1, false, ldsB(sW2, nt, 32, lane),
                                                       (short)0, c, false, false);
#pragma unroll
            for (int r = 0; r < 8; ++r)
                sO[wave][(r + hi * 8) * D_ + nt * 16 + colc] = c[r];
        }
    }
    __syncthreads();

    // LayerNorm: lanes 0..15 each normalize one row and store
    if (lane < 16) {
        int n = node0 + lane;
        if (n < nN) {
            const float* rp = &sO[wave][lane * D_];
            float mu = 0.f;
#pragma unroll
            for (int c0 = 0; c0 < D_; ++c0) mu += rp[c0];
            mu *= (1.0f / D_);
            float var = 0.f;
#pragma unroll
            for (int c0 = 0; c0 < D_; ++c0) { float d = rp[c0] - mu; var += d * d; }
            var *= (1.0f / D_);
            float rs = rsqrtf(var + 1e-5f);
            float4* po = (float4*)(xout + (size_t)n * D_);
#pragma unroll
            for (int q = 0; q < 16; ++q) {
                float4 o;
                int c0 = q * 4;
                o.x = (rp[c0 + 0] - mu) * rs * sG[c0 + 0] + sBt[c0 + 0];
                o.y = (rp[c0 + 1] - mu) * rs * sG[c0 + 1] + sBt[c0 + 1];
                o.z = (rp[c0 + 2] - mu) * rs * sG[c0 + 2] + sBt[c0 + 2];
                o.w = (rp[c0 + 3] - mu) * rs * sG[c0 + 3] + sBt[c0 + 3];
                po[q] = o;
            }
        }
    }
}

// ---------------- gate MLP (64->32 SiLU ->1) + segment max ----------------
__global__ __launch_bounds__(256) void k_gate(const float* __restrict__ x,
                                              const int* __restrict__ batch,
                                              const float* __restrict__ gw1,
                                              const float* __restrict__ gb1,
                                              const float* __restrict__ gw2,
                                              const float* __restrict__ gb2,
                                              float* __restrict__ gate,
                                              unsigned* __restrict__ mbuf, int nN) {
    __shared__ float sW[D_ * HID];
    __shared__ float sb1[HID];
    __shared__ float sw2[HID];
    __shared__ float sb2v;
    int tid = threadIdx.x;
    for (int i = tid; i < D_ * HID; i += 256) sW[i] = gw1[i];
    if (tid < HID) { sb1[tid] = gb1[tid]; sw2[tid] = gw2[tid]; }
    if (tid == 0) sb2v = gb2[0];
    __syncthreads();
    int n = blockIdx.x * 256 + tid;
    if (n >= nN) return;
    float acc[HID];
#pragma unroll
    for (int j = 0; j < HID; ++j) acc[j] = sb1[j];
    const float4* px = (const float4*)(x + (size_t)n * D_);
#pragma unroll 4
    for (int k4 = 0; k4 < 16; ++k4) {
        float4 v = px[k4];
        const float* w0 = &sW[(k4 * 4 + 0) * HID];
        const float* w1 = &sW[(k4 * 4 + 1) * HID];
        const float* w2 = &sW[(k4 * 4 + 2) * HID];
        const float* w3 = &sW[(k4 * 4 + 3) * HID];
#pragma unroll
        for (int j = 0; j < HID; ++j)
            acc[j] += v.x * w0[j] + v.y * w1[j] + v.z * w2[j] + v.w * w3[j];
    }
    float g = sb2v;
#pragma unroll
    for (int j = 0; j < HID; ++j) {
        float h = acc[j];
        h = h * (1.0f / (1.0f + __expf(-h)));
        g += h * sw2[j];
    }
    gate[n] = g;
    atomicMax(&mbuf[batch[n]], encf(g));
}

// ---------------- e = exp(gate - max), denom accumulation ----------------
__global__ __launch_bounds__(256) void k_exp(const float* __restrict__ gate,
                                             const int* __restrict__ batch,
                                             const unsigned* __restrict__ mbuf,
                                             float* __restrict__ ebuf,
                                             float* __restrict__ denom, int nN) {
    int n = blockIdx.x * 256 + threadIdx.x;
    if (n >= nN) return;
    int b = batch[n];
    float ev = __expf(gate[n] - decf(mbuf[b]));
    ebuf[n] = ev;
    unsafeAtomicAdd(&denom[b], ev);
}

// ---------------- out[g] += (e/denom) * x[n] ----------------
__global__ __launch_bounds__(256) void k_accum(const float* __restrict__ x,
                                               const int* __restrict__ batch,
                                               const float* __restrict__ ebuf,
                                               const float* __restrict__ denom,
                                               float* __restrict__ out, int total, int nN) {
    int idx = blockIdx.x * 256 + threadIdx.x;
    if (idx >= total) return;
    int n = idx >> 4, q = idx & 15;
    int b = batch[n];
    float a = ebuf[n] / denom[b];
    float4 v = ((const float4*)(x + (size_t)n * D_))[q];
    float* o = out + (size_t)b * D_ + q * 4;
    unsafeAtomicAdd(o + 0, a * v.x);
    unsafeAtomicAdd(o + 1, a * v.y);
    unsafeAtomicAdd(o + 2, a * v.z);
    unsafeAtomicAdd(o + 3, a * v.w);
}

// ---------------- host launcher ----------------
extern "C" void kernel_launch(void* const* d_in, const int* in_sizes, int n_in,
                              void* d_out, int out_size, void* d_ws, size_t ws_size,
                              hipStream_t stream) {
    const float* x    = (const float*)d_in[0];
    const int*   ei   = (const int*)d_in[1];
    const int*   batch= (const int*)d_in[2];
    const float* W1   = (const float*)d_in[3];
    const float* b1   = (const float*)d_in[4];
    const float* W2   = (const float*)d_in[5];
    const float* b2   = (const float*)d_in[6];
    const float* eps  = (const float*)d_in[7];
    const float* lng  = (const float*)d_in[8];
    const float* lnb  = (const float*)d_in[9];
    const float* gw1  = (const float*)d_in[10];
    const float* gb1  = (const float*)d_in[11];
    const float* gw2  = (const float*)d_in[12];
    const float* gb2  = (const float*)d_in[13];
    float* out = (float*)d_out;

    const int nN = in_sizes[0] / D_;
    const int nE = in_sizes[1] / 2;
    const int nL = in_sizes[7];
    const int nG = out_size / D_;
    const int* src = ei;
    const int* dst = ei + nE;

    char* p = (char*)d_ws;
    auto alloc = [&](size_t bytes) -> char* {
        char* r = p;
        p += (bytes + 255) & ~(size_t)255;
        return r;
    };
    float* x_cur = (float*)alloc((size_t)nN * D_ * 4);
    float* agg   = (float*)alloc((size_t)nN * D_ * 4);
    unsigned short* w1t = (unsigned short*)alloc((size_t)nL * D_ * D_ * 2);
    unsigned short* w2t = (unsigned short*)alloc((size_t)nL * D_ * D_ * 2);
    float* gate  = (float*)alloc((size_t)nN * 4);
    float* ebuf  = (float*)alloc((size_t)nN * 4);
    unsigned* mbuf = (unsigned*)alloc((size_t)nG * 4);
    float* denom = (float*)alloc((size_t)nG * 4);

    {
        int total = nL * D_ * D_;
        k_convert<<<(total + 255) / 256, 256, 0, stream>>>(W1, W2, w1t, w2t, total);
    }

    for (int l = 0; l < nL; ++l) {
        const float* xin = (l == 0) ? x : x_cur;
        hipMemsetAsync(agg, 0, (size_t)nN * D_ * 4, stream);
        int st = nE * 16;
        k_scatter<<<(st + 255) / 256, 256, 0, stream>>>(xin, src, dst, agg, nE);
        k_gin_mlp<<<(nN + 63) / 64, 128, 0, stream>>>(xin, agg, w1t, w2t, b1, b2, eps,
                                                      lng, lnb, x_cur, l, nN);
    }

    hipMemsetAsync(mbuf, 0, (size_t)nG * 4, stream);
    hipMemsetAsync(denom, 0, (size_t)nG * 4, stream);
    hipMemsetAsync(out, 0, (size_t)nG * D_ * 4, stream);
    k_gate<<<(nN + 255) / 256, 256, 0, stream>>>(x_cur, batch, gw1, gb1, gw2, gb2, gate, mbuf, nN);
    k_exp<<<(nN + 255) / 256, 256, 0, stream>>>(gate, batch, mbuf, ebuf, denom, nN);
    int at = nN * 16;
    k_accum<<<(at + 255) / 256, 256, 0, stream>>>(x_cur, batch, ebuf, denom, out, at, nN);
}
